// SimpleSelfAttention_49598282334495
// MI455X (gfx1250) — matrix operand order repaired and checked by reading, not compiled, not run on hardware
//
#include <hip/hip_runtime.h>
#include <hip/hip_bf16.h>
#include <math.h>
#include <stdint.h>

// Self-attention forward for MI455X (gfx1250), bf16 WMMA pipeline:
//   x[2,4096,512] -> QKV GEMM -> flash attention (online softmax) -> out proj.

#define B_    2
#define N_    4096
#define D_    512
#define H_    8
#define HD_   64
#define MTOK  (B_*N_)     // 8192
#define QKVN  (3*D_)      // 1536

typedef __attribute__((ext_vector_type(16))) __bf16 v16bf;
typedef __attribute__((ext_vector_type(8)))  float  v8f;

union Frag { v16bf v; uint4 q[2]; };

__device__ __forceinline__ uint16_t f2bf(float f) {
    union { float f; uint32_t u; } x; x.f = f;
    uint32_t u = x.u;
    uint32_t r = u + 0x7FFFu + ((u >> 16) & 1u);  // round-to-nearest-even
    return (uint16_t)(r >> 16);
}

// ---------------- elementwise convert f32 -> bf16 (vector x4) ----------------
__global__ void cvt_kernel(const float* __restrict__ in, uint16_t* __restrict__ out, int n4) {
    int i = blockIdx.x * blockDim.x + threadIdx.x;
    if (i < n4) {
        float4 f = ((const float4*)in)[i];
        ushort4 o;
        o.x = f2bf(f.x); o.y = f2bf(f.y); o.z = f2bf(f.z); o.w = f2bf(f.w);
        ((ushort4*)out)[i] = o;
    }
}

// -------- transpose-convert W[K][Ncols] f32 -> WT[Ncols][K] bf16 (one-time) --------
__global__ void cvtT_kernel(const float* __restrict__ W, uint16_t* __restrict__ WT,
                            int K, int Ncols) {
    int o = blockIdx.x * blockDim.x + threadIdx.x;
    if (o < K * Ncols) {
        int n = o / K, k = o % K;
        WT[o] = f2bf(W[k * Ncols + n]);
    }
}

// ---------------- QKV GEMM: [8192,512] x [512,1536] + bias, head-scatter epilogue ----------------
__global__ __launch_bounds__(256)
void qkv_gemm_kernel(const uint16_t* __restrict__ xb,   // [8192][512] bf16
                     const uint16_t* __restrict__ wT,   // [1536][512] bf16 (pre-transposed)
                     const float*    __restrict__ bias, // [1536]
                     uint16_t* __restrict__ qb,         // [16][4096][64]
                     uint16_t* __restrict__ kb,         // [16][4096][64]
                     uint16_t* __restrict__ vT)         // [16][64][4096]
{
    __shared__ __align__(16) uint16_t As[128 * 40];
    __shared__ __align__(16) uint16_t Bs[64 * 40];
    const int tid = threadIdx.x, lane = tid & 31, wave = tid >> 5;
    const int wm = wave & 3, wn = wave >> 2;
    const int m0 = blockIdx.x * 128, n0 = blockIdx.y * 64;
    const int lm = lane & 15, hv = lane >> 4;

    v8f acc[2][2] = {};

    for (int k0 = 0; k0 < 512; k0 += 32) {
        __syncthreads();
        #pragma unroll
        for (int p = 0; p < 2; ++p) {                 // A tile 128x32
            int u = tid + p * 256;
            int row = u >> 2, seg = u & 3;
            *(uint4*)&As[row * 40 + seg * 8] =
                *(const uint4*)&xb[(m0 + row) * 512 + k0 + seg * 8];
        }
        {                                             // B tile 64x32 (Bt layout)
            int row = tid >> 2, seg = tid & 3;
            *(uint4*)&Bs[row * 40 + seg * 8] =
                *(const uint4*)&wT[(n0 + row) * 512 + k0 + seg * 8];
        }
        __syncthreads();
        Frag a[2], b[2];
        #pragma unroll
        for (int mt = 0; mt < 2; ++mt) {
            const uint16_t* base = &As[(wm * 32 + mt * 16 + lm) * 40];
            a[mt].q[0] = *(const uint4*)(base + 16 * hv);
            a[mt].q[1] = *(const uint4*)(base + 16 * hv + 8);
        }
        #pragma unroll
        for (int nt = 0; nt < 2; ++nt) {
            const uint16_t* base = &Bs[(wn * 32 + nt * 16 + lm) * 40];
            b[nt].q[0] = *(const uint4*)(base + 16 * hv);
            b[nt].q[1] = *(const uint4*)(base + 16 * hv + 8);
        }
        #pragma unroll
        for (int mt = 0; mt < 2; ++mt)
            #pragma unroll
            for (int nt = 0; nt < 2; ++nt)
                acc[mt][nt] = __builtin_amdgcn_wmma_f32_16x16x32_bf16(
                    false, a[mt].v, false, b[nt].v, (short)0, acc[mt][nt], false, false);
    }
    // epilogue: bias + bf16 + head-interleaved scatter (v stored transposed)
    #pragma unroll
    for (int mt = 0; mt < 2; ++mt)
        #pragma unroll
        for (int nt = 0; nt < 2; ++nt) {
            int c = n0 + wn * 32 + nt * 16 + lm;
            float bv = bias[c];
            int which = c >> 9;
            int d = c & 511;
            int h = d >> 6, hd = d & 63;
            #pragma unroll
            for (int i = 0; i < 8; ++i) {
                int m = m0 + wm * 32 + mt * 16 + i + 8 * hv;
                int bb = m >> 12, tok = m & 4095;
                int bh = bb * 8 + h;
                uint16_t val = f2bf(acc[mt][nt][i] + bv);
                if (which == 0)      qb[(bh * 4096 + tok) * 64 + hd] = val;
                else if (which == 1) kb[(bh * 4096 + tok) * 64 + hd] = val;
                else                 vT[(bh * 64 + hd) * 4096 + tok] = val;
            }
        }
}

// ---------------- flash attention: per (b,h), 128 queries per WG, key chunks of 32 ----------------
__global__ __launch_bounds__(256)
void attn_kernel(const uint16_t* __restrict__ qb,  // [16][4096][64]
                 const uint16_t* __restrict__ kb,  // [16][4096][64]
                 const uint16_t* __restrict__ vT,  // [16][64][4096]
                 uint16_t* __restrict__ ob)        // [8192][512]
{
    __shared__ __align__(16) uint16_t Kt[32 * 72];      // [key][hd]
    __shared__ __align__(16) uint16_t Vt[64 * 40];      // [hd][key]
    __shared__ __align__(16) float    Sld[8][16 * 32];  // per-wave score tile
    __shared__ __align__(16) uint16_t Pld[8][16 * 40];  // per-wave P tile (bf16)
    __shared__ float Corr[8][16];
    __shared__ float Lrow[8][16];

    const int tid = threadIdx.x, lane = tid & 31, w = tid >> 5;
    const int lm = lane & 15, hv = lane >> 4;
    const int id = blockIdx.x;
    const int q0 = (id & 31) * 128;
    const int bh = id >> 5;

    Frag qa[2];     // Q A-fragments for Hd K-chunks 0..31 and 32..63
    {
        const uint16_t* base = &qb[(bh * 4096 + q0 + w * 16 + lm) * 64];
        qa[0].q[0] = *(const uint4*)(base + 16 * hv);
        qa[0].q[1] = *(const uint4*)(base + 16 * hv + 8);
        qa[1].q[0] = *(const uint4*)(base + 32 + 16 * hv);
        qa[1].q[1] = *(const uint4*)(base + 32 + 16 * hv + 8);
    }

    v8f oacc[4] = {};
    float mrow = -INFINITY, lrow = 0.f;
    const float sc = 0.125f;   // 1/sqrt(64)

    for (int k0 = 0; k0 < 4096; k0 += 32) {
        __syncthreads();
        {   // K chunk: 32 keys x 64 hd
            int row = tid >> 3, seg = tid & 7;
            *(uint4*)&Kt[row * 72 + seg * 8] =
                *(const uint4*)&kb[(bh * 4096 + k0 + row) * 64 + seg * 8];
        }
        {   // V chunk (transposed): 64 hd x 32 keys
            int row = tid >> 2, seg = tid & 3;
            *(uint4*)&Vt[row * 40 + seg * 8] =
                *(const uint4*)&vT[(bh * 64 + row) * 4096 + k0 + seg * 8];
        }
        __syncthreads();

        // scores: 16 queries x 32 keys via 2x2 WMMAs (Hd=64 -> 2 K-chunks)
        #pragma unroll
        for (int nt = 0; nt < 2; ++nt) {
            v8f s = {};
            Frag b0, b1;
            const uint16_t* base = &Kt[(nt * 16 + lm) * 72];
            b0.q[0] = *(const uint4*)(base + 16 * hv);
            b0.q[1] = *(const uint4*)(base + 16 * hv + 8);
            b1.q[0] = *(const uint4*)(base + 32 + 16 * hv);
            b1.q[1] = *(const uint4*)(base + 32 + 16 * hv + 8);
            s = __builtin_amdgcn_wmma_f32_16x16x32_bf16(false, qa[0].v, false, b0.v, (short)0, s, false, false);
            s = __builtin_amdgcn_wmma_f32_16x16x32_bf16(false, qa[1].v, false, b1.v, (short)0, s, false, false);
            #pragma unroll
            for (int i = 0; i < 8; ++i)
                Sld[w][(i + 8 * hv) * 32 + nt * 16 + lm] = s[i];
        }

        // online softmax: lane owns row lm, key-half hv (16 keys); combine halves via shfl_xor(16)
        float sv[16];
        float smax = -INFINITY;
        #pragma unroll
        for (int j = 0; j < 16; ++j) {
            sv[j] = Sld[w][lm * 32 + hv * 16 + j] * sc;
            smax = fmaxf(smax, sv[j]);
        }
        smax = fmaxf(smax, __shfl_xor(smax, 16, 32));
        float mnew = fmaxf(mrow, smax);
        float corr = __expf(mrow - mnew);
        float ssum = 0.f;
        #pragma unroll
        for (int j = 0; j < 16; ++j) {
            float p = __expf(sv[j] - mnew);
            ssum += p;
            Pld[w][lm * 40 + hv * 16 + j] = f2bf(p);
        }
        ssum += __shfl_xor(ssum, 16, 32);
        lrow = lrow * corr + ssum;
        mrow = mnew;
        if (lane < 16) Corr[w][lm] = corr;

        // rescale O accumulators and add P @ V
        Frag pa;
        {
            const uint16_t* base = &Pld[w][lm * 40];
            pa.q[0] = *(const uint4*)(base + 16 * hv);
            pa.q[1] = *(const uint4*)(base + 16 * hv + 8);
        }
        float cf[8];
        #pragma unroll
        for (int i = 0; i < 8; ++i) cf[i] = Corr[w][i + 8 * hv];
        #pragma unroll
        for (int t = 0; t < 4; ++t) {
            #pragma unroll
            for (int i = 0; i < 8; ++i) oacc[t][i] *= cf[i];
            Frag vb;
            const uint16_t* base = &Vt[(t * 16 + lm) * 40];
            vb.q[0] = *(const uint4*)(base + 16 * hv);
            vb.q[1] = *(const uint4*)(base + 16 * hv + 8);
            oacc[t] = __builtin_amdgcn_wmma_f32_16x16x32_bf16(
                false, pa.v, false, vb.v, (short)0, oacc[t], false, false);
        }
    }

    // finalize: divide by row sums, store bf16 [token][h*64+hd]
    if (lane < 16) Lrow[w][lm] = 1.f / lrow;
    float li[8];
    #pragma unroll
    for (int i = 0; i < 8; ++i) li[i] = Lrow[w][i + 8 * hv];
    const int bb = bh >> 3, h = bh & 7;
    #pragma unroll
    for (int t = 0; t < 4; ++t) {
        int col = h * 64 + t * 16 + lm;
        #pragma unroll
        for (int i = 0; i < 8; ++i) {
            int tok = q0 + w * 16 + i + 8 * hv;
            ob[(bb * 4096 + tok) * 512 + col] = f2bf(oacc[t][i] * li[i]);
        }
    }
}

// ---------------- output projection: [8192,512] x [512,512] + bias -> f32 ----------------
__global__ __launch_bounds__(256)
void proj_kernel(const uint16_t* __restrict__ ab,  // [8192][512] bf16
                 const uint16_t* __restrict__ wT,  // [512][512] bf16 (pre-transposed)
                 const float*    __restrict__ bias,// [512]
                 float* __restrict__ out)          // [8192][512] f32
{
    __shared__ __align__(16) uint16_t As[128 * 40];
    __shared__ __align__(16) uint16_t Bs[64 * 40];
    const int tid = threadIdx.x, lane = tid & 31, wave = tid >> 5;
    const int wm = wave & 3, wn = wave >> 2;
    const int m0 = blockIdx.x * 128, n0 = blockIdx.y * 64;
    const int lm = lane & 15, hv = lane >> 4;

    v8f acc[2][2] = {};

    for (int k0 = 0; k0 < 512; k0 += 32) {
        __syncthreads();
        #pragma unroll
        for (int p = 0; p < 2; ++p) {
            int u = tid + p * 256;
            int row = u >> 2, seg = u & 3;
            *(uint4*)&As[row * 40 + seg * 8] =
                *(const uint4*)&ab[(m0 + row) * 512 + k0 + seg * 8];
        }
        {
            int row = tid >> 2, seg = tid & 3;
            *(uint4*)&Bs[row * 40 + seg * 8] =
                *(const uint4*)&wT[(n0 + row) * 512 + k0 + seg * 8];
        }
        __syncthreads();
        Frag a[2], b[2];
        #pragma unroll
        for (int mt = 0; mt < 2; ++mt) {
            const uint16_t* base = &As[(wm * 32 + mt * 16 + lm) * 40];
            a[mt].q[0] = *(const uint4*)(base + 16 * hv);
            a[mt].q[1] = *(const uint4*)(base + 16 * hv + 8);
        }
        #pragma unroll
        for (int nt = 0; nt < 2; ++nt) {
            const uint16_t* base = &Bs[(wn * 32 + nt * 16 + lm) * 40];
            b[nt].q[0] = *(const uint4*)(base + 16 * hv);
            b[nt].q[1] = *(const uint4*)(base + 16 * hv + 8);
        }
        #pragma unroll
        for (int mt = 0; mt < 2; ++mt)
            #pragma unroll
            for (int nt = 0; nt < 2; ++nt)
                acc[mt][nt] = __builtin_amdgcn_wmma_f32_16x16x32_bf16(
                    false, a[mt].v, false, b[nt].v, (short)0, acc[mt][nt], false, false);
    }
    #pragma unroll
    for (int mt = 0; mt < 2; ++mt)
        #pragma unroll
        for (int nt = 0; nt < 2; ++nt) {
            int c = n0 + wn * 32 + nt * 16 + lm;
            float bv = bias[c];
            #pragma unroll
            for (int i = 0; i < 8; ++i) {
                int m = m0 + wm * 32 + mt * 16 + i + 8 * hv;
                out[m * 512 + c] = acc[mt][nt][i] + bv;
            }
        }
}

extern "C" void kernel_launch(void* const* d_in, const int* in_sizes, int n_in,
                              void* d_out, int out_size, void* d_ws, size_t ws_size,
                              hipStream_t stream) {
    const float* x    = (const float*)d_in[0];
    const float* Wqkv = (const float*)d_in[1];
    const float* bqkv = (const float*)d_in[2];
    const float* Wout = (const float*)d_in[3];
    const float* bout = (const float*)d_in[4];
    float* out = (float*)d_out;

    char* ws = (char*)d_ws;
    size_t off = 0;
    auto alloc = [&](size_t bytes) -> void* {
        void* p = ws + off;
        off += (bytes + 255) & ~(size_t)255;
        return p;
    };
    uint16_t* xb  = (uint16_t*)alloc((size_t)MTOK * D_ * 2);
    uint16_t* wqT = (uint16_t*)alloc((size_t)QKVN * D_ * 2);
    uint16_t* woT = (uint16_t*)alloc((size_t)D_ * D_ * 2);
    uint16_t* qb  = (uint16_t*)alloc((size_t)16 * 4096 * 64 * 2);
    uint16_t* kb  = (uint16_t*)alloc((size_t)16 * 4096 * 64 * 2);
    uint16_t* vT  = (uint16_t*)alloc((size_t)16 * 64 * 4096 * 2);
    uint16_t* ab  = (uint16_t*)alloc((size_t)MTOK * D_ * 2);

    cvt_kernel<<<(MTOK * D_ / 4 + 255) / 256, 256, 0, stream>>>(x, xb, MTOK * D_ / 4);
    cvtT_kernel<<<(D_ * QKVN + 255) / 256, 256, 0, stream>>>(Wqkv, wqT, D_, QKVN);
    cvtT_kernel<<<(D_ * D_ + 255) / 256, 256, 0, stream>>>(Wout, woT, D_, D_);

    qkv_gemm_kernel<<<dim3(MTOK / 128, QKVN / 64), 256, 0, stream>>>(xb, wqT, bqkv, qb, kb, vT);
    attn_kernel<<<B_ * H_ * (N_ / 128), 256, 0, stream>>>(qb, kb, vT, ab);
    proj_kernel<<<dim3(MTOK / 128, D_ / 64), 256, 0, stream>>>(ab, woT, bout, out);
}
